// TransformerDecoder_24876450579329
// MI455X (gfx1250) — matrix operand order repaired and checked by reading, not compiled, not run on hardware
//
#include <hip/hip_runtime.h>
#include <cstddef>
#include <cstdint>

// ---------------------------------------------------------------------------
// Transformer decoder for MI455X (gfx1250, wave32, WMMA).
// Shapes: L=6, D=1024, H=16, DH=64, F=4096, B=8, S=M=512.
// - GEMMs: v_wmma_f32_16x16x32_f16, f16 weights pre-transposed once per call,
//   B tiles staged via global_load_async_to_lds_b128 (double-buffered,
//   ASYNCcnt-pipelined), A tiles register-pipelined fp32->f16.
// - Attention: flash-style, LDS-staged K/V, online softmax, 8 WMMAs/step.
// ---------------------------------------------------------------------------

typedef _Float16 v16h __attribute__((ext_vector_type(16)));
typedef _Float16 v8h  __attribute__((ext_vector_type(8)));
typedef _Float16 v4h  __attribute__((ext_vector_type(4)));
typedef float    v8f  __attribute__((ext_vector_type(8)));

static constexpr int kB  = 8;
static constexpr int kS  = 512;
static constexpr int kD  = 1024;
static constexpr int kH  = 16;
static constexpr int kDH = 64;
static constexpr int kF  = 4096;
static constexpr int kRows = kB * kS;          // 4096 token rows

__device__ __forceinline__ v8f wmma_f16(v16h a, v16h b, v8f c) {
  return __builtin_amdgcn_wmma_f32_16x16x32_f16(false, a, false, b, (short)0, c,
                                                false, false);
}

// K-index of half-element e in the 16-bit A-matrix 16x32 fragment (ISA 7.12.2)
__device__ __forceinline__ int a_koff(int hi, int e) {
  return e + 8 * ((e >= 8) + hi);
}

// Async global->LDS 128-bit copy (ASYNCcnt).  lds_off is a wave-relative LDS
// byte address (low 32 bits of the generic shared pointer), gaddr is global.
__device__ __forceinline__ void async_ld_b128(uint32_t lds_off,
                                              const void* gaddr) {
  asm volatile("global_load_async_to_lds_b128 %0, %1, off"
               :: "v"(lds_off), "v"(gaddr)
               : "memory");
}

// ---------------------------------------------------------------------------
// Weight transpose+convert: Wt[N][K] f16  <-  W[K][N] fp32  (one pass, tiled)
// ---------------------------------------------------------------------------
__global__ void __launch_bounds__(256)
wcvt_t(const float* __restrict__ W, _Float16* __restrict__ Wt, int K, int N) {
  __shared__ float tile[32][33];
  const int tx = threadIdx.x & 31;
  const int ty = (threadIdx.x >> 5) * 4;       // 8 waves -> 4 rows each
  const int n0 = blockIdx.x * 32;
  const int k0 = blockIdx.y * 32;
#pragma unroll
  for (int i = 0; i < 4; i++)
    tile[ty + i][tx] = W[(size_t)(k0 + ty + i) * N + n0 + tx];
  __syncthreads();
#pragma unroll
  for (int i = 0; i < 4; i++)
    Wt[(size_t)(n0 + ty + i) * K + k0 + tx] = (_Float16)tile[tx][ty + i];
}

// ---------------------------------------------------------------------------
// GEMM: C[Mr x N] = A[Mr x K] @ Wt[N x K]^T + bias (+residual) (optional relu)
// Block = 256 threads (8 waves), 128x128 tile; wave -> 32x64 (2x4 WMMA tiles).
// B tiles: async DMA into double-buffered LDS; A tiles: register pipeline.
// ---------------------------------------------------------------------------
template <bool RESID, bool RELU>
__global__ void __launch_bounds__(256)
gemm_f16(const float* __restrict__ A, const _Float16* __restrict__ Wt,
         const float* __restrict__ bias, const float* __restrict__ resid,
         float* __restrict__ C, int Mr, int K, int N) {
  __shared__ __align__(16) _Float16 As[128 * 40];     // row-major, padded
  __shared__ __align__(16) _Float16 Bs[2][128 * 40];  // Bs[b][n*40 + k]

  const int t    = threadIdx.x;
  const int lane = t & 31;
  const int wv   = t >> 5;
  const int hi   = lane >> 4;          // 0 | 1 (half-wave)
  const int ln   = lane & 15;
  const int wr   = (wv & 3) * 32;      // wave row offset in 128-tile
  const int wc   = (wv >> 2) * 64;     // wave col offset
  const int row0 = blockIdx.y * 128;
  const int col0 = blockIdx.x * 128;

  const v8f vzero = {};
  v8f acc[2][4];
#pragma unroll
  for (int i = 0; i < 2; i++)
#pragma unroll
    for (int j = 0; j < 4; j++) acc[i][j] = vzero;

  const int lr  = t >> 1;          // A-load: row 0..127
  const int lcb = (t & 1) * 16;    // A-load: col base {0,16}

  const float* aptr = A + (size_t)(row0 + lr) * K + lcb;

  // B staging: thread t owns row n = t/2, 32-byte half-row (t&1)
  const int bn  = t >> 1;
  const int bco = (t & 1) * 16;    // half offset within 32-half row
  const _Float16* wrow = Wt + (size_t)(col0 + bn) * K + bco;
  const uint32_t bofs[2] = {
      (uint32_t)(uintptr_t)&Bs[0][bn * 40 + bco],
      (uint32_t)(uintptr_t)&Bs[1][bn * 40 + bco]};
  auto issueW = [&](int k0, int pb) {
    const _Float16* src = wrow + k0;
    async_ld_b128(bofs[pb], src);
    async_ld_b128(bofs[pb] + 16, src + 8);
  };

  float4 aR[4];
  auto gloadA = [&](int k0) {
    const float4* as = reinterpret_cast<const float4*>(aptr + k0);
#pragma unroll
    for (int j = 0; j < 4; j++) aR[j] = as[j];
  };

  issueW(0, 0);
  gloadA(0);
  int pb = 0;
  for (int k0 = 0; k0 < K; k0 += 32, pb ^= 1) {
    // ---- commit staged A registers to LDS (fp32 -> f16) ----
#pragma unroll
    for (int j = 0; j < 4; j++) {
      float4 f = aR[j];
      v4h h = {(_Float16)f.x, (_Float16)f.y, (_Float16)f.z, (_Float16)f.w};
      *reinterpret_cast<v4h*>(&As[lr * 40 + lcb + j * 4]) = h;
    }
    // ---- kick DMA for next B tile, then wait for current one ----
    if (k0 + 32 < K) {
      issueW(k0 + 32, pb ^ 1);
      asm volatile("s_wait_asynccnt 0x2" ::: "memory");
    } else {
      asm volatile("s_wait_asynccnt 0x0" ::: "memory");
    }
    __syncthreads();

    // ---- pipeline A: fetch next tile, prefetch tile after that ----
    if (k0 + 32 < K) gloadA(k0 + 32);
    if (k0 + 64 < K) __builtin_prefetch(aptr + k0 + 64, 0, 1);

    // ---- fragments + 8 WMMAs ----
    v16h af[2], bf[4];
#pragma unroll
    for (int mt = 0; mt < 2; mt++) {
      const int r  = wr + mt * 16 + ln;
      const int c0 = hi ? 8 : 0;
      v8h lo = *reinterpret_cast<const v8h*>(&As[r * 40 + c0]);
      v8h hh = *reinterpret_cast<const v8h*>(&As[r * 40 + c0 + 16]);
#pragma unroll
      for (int e = 0; e < 8; e++) { af[mt][e] = lo[e]; af[mt][8 + e] = hh[e]; }
    }
#pragma unroll
    for (int nt = 0; nt < 4; nt++) {
      const int n  = wc + nt * 16 + ln;
      const int c0 = hi ? 8 : 0;
      v8h lo = *reinterpret_cast<const v8h*>(&Bs[pb][n * 40 + c0]);
      v8h hh = *reinterpret_cast<const v8h*>(&Bs[pb][n * 40 + c0 + 16]);
#pragma unroll
      for (int e = 0; e < 8; e++) { bf[nt][e] = lo[e]; bf[nt][8 + e] = hh[e]; }
    }
#pragma unroll
    for (int mt = 0; mt < 2; mt++)
#pragma unroll
      for (int nt = 0; nt < 4; nt++)
        acc[mt][nt] = wmma_f16(af[mt], bf[nt], acc[mt][nt]);
    __syncthreads();
  }

  // ---- branchless epilogue: bias (+resid) (+relu) ----
#pragma unroll
  for (int mt = 0; mt < 2; mt++) {
#pragma unroll
    for (int nt = 0; nt < 4; nt++) {
      const int col = col0 + wc + nt * 16 + ln;
      const float bval = bias[col];
      const size_t base = (size_t)(row0 + wr + mt * 16 + 8 * hi) * N + col;
#pragma unroll
      for (int r = 0; r < 8; r++) {
        float v = acc[mt][nt][r] + bval;
        if (RESID) v += resid[base + (size_t)r * N];
        if (RELU) v = fmaxf(v, 0.0f);
        C[base + (size_t)r * N] = v;
      }
    }
  }
}

// ---------------------------------------------------------------------------
// Flash attention: block = 8 waves sharing (b,h); each wave owns a 16-row q
// tile; K/V tiles (32 keys) are cooperatively staged into LDS as f16 once per
// block-step.  scores: 2 WMMAs/tile (DH=64 split 2x32), ctx: 4 WMMAs/step.
// ---------------------------------------------------------------------------
__global__ void __launch_bounds__(256)
attn_flash(const float* __restrict__ Q, const float* __restrict__ Kb,
           const float* __restrict__ Vb, const float* __restrict__ mask,
           float* __restrict__ O, int Skv) {
  __shared__ __align__(16) _Float16 Ks[32 * 72];      // [key][d], padded
  __shared__ __align__(16) _Float16 Vt[64 * 40];      // [d][key], padded
  __shared__ __align__(16) _Float16 Pl[8][16 * 40];   // per-wave P transpose

  const int t    = threadIdx.x;
  const int lane = t & 31;
  const int wv   = t >> 5;
  const int hi   = lane >> 4;
  const int ln   = lane & 15;
  const int bh   = blockIdx.y;
  const int b    = bh >> 4;        // / H
  const int h    = bh & 15;        // % H
  const int q0   = blockIdx.x * 128 + wv * 16;

  const size_t qbase  = ((size_t)b * kS + q0) * kD + (size_t)h * kDH;
  const size_t kvbase = (size_t)b * Skv * kD + (size_t)h * kDH;

  // staging assignment: thread t handles key j = t/8, d = (t%8)*8 .. +7
  const int sj = t >> 3;
  const int sd = (t & 7) * 8;

  // Q fragments (A-layout), scale 1/sqrt(DH) folded in
  v16h qf[2];
#pragma unroll
  for (int c = 0; c < 2; c++) {
    const float* qp = Q + qbase + (size_t)ln * kD + c * 32;
#pragma unroll
    for (int e = 0; e < 16; e++)
      qf[c][e] = (_Float16)(0.125f * qp[a_koff(hi, e)]);
  }

  const v8f vzero = {};
  v8f ctx[4];
#pragma unroll
  for (int i = 0; i < 4; i++) ctx[i] = vzero;
  float mrow[8], lrow[8];
#pragma unroll
  for (int r = 0; r < 8; r++) { mrow[r] = -3.0e38f; lrow[r] = 0.0f; }

  for (int kb0 = 0; kb0 < Skv; kb0 += 32) {
    // ---- cooperative stage of K and V tiles (32 keys x 64 d) into LDS ----
    {
      const float* kp = Kb + kvbase + (size_t)(kb0 + sj) * kD + sd;
      const float* vp = Vb + kvbase + (size_t)(kb0 + sj) * kD + sd;
      const float4 k0 = reinterpret_cast<const float4*>(kp)[0];
      const float4 k1 = reinterpret_cast<const float4*>(kp)[1];
      const float4 u0 = reinterpret_cast<const float4*>(vp)[0];
      const float4 u1 = reinterpret_cast<const float4*>(vp)[1];
      v4h h0 = {(_Float16)k0.x, (_Float16)k0.y, (_Float16)k0.z, (_Float16)k0.w};
      v4h h1 = {(_Float16)k1.x, (_Float16)k1.y, (_Float16)k1.z, (_Float16)k1.w};
      *reinterpret_cast<v4h*>(&Ks[sj * 72 + sd])     = h0;
      *reinterpret_cast<v4h*>(&Ks[sj * 72 + sd + 4]) = h1;
      Vt[(sd + 0) * 40 + sj] = (_Float16)u0.x;
      Vt[(sd + 1) * 40 + sj] = (_Float16)u0.y;
      Vt[(sd + 2) * 40 + sj] = (_Float16)u0.z;
      Vt[(sd + 3) * 40 + sj] = (_Float16)u0.w;
      Vt[(sd + 4) * 40 + sj] = (_Float16)u1.x;
      Vt[(sd + 5) * 40 + sj] = (_Float16)u1.y;
      Vt[(sd + 6) * 40 + sj] = (_Float16)u1.z;
      Vt[(sd + 7) * 40 + sj] = (_Float16)u1.w;
      if (kb0 + 32 < Skv) {
        __builtin_prefetch(kp + 32 * kD, 0, 1);
        __builtin_prefetch(vp + 32 * kD, 0, 1);
      }
    }
    __syncthreads();

    // ---- K^T fragments from LDS, scores = Q @ K^T (two 16x16 tiles) ----
    v8f s[2];
#pragma unroll
    for (int jt = 0; jt < 2; jt++) {
      v16h kf0, kf1;
      const int j = jt * 16 + ln;
#pragma unroll
      for (int c = 0; c < 2; c++) {
        const int d0 = c * 32 + (hi ? 8 : 0);
        v8h lo = *reinterpret_cast<const v8h*>(&Ks[j * 72 + d0]);
        v8h hh = *reinterpret_cast<const v8h*>(&Ks[j * 72 + d0 + 16]);
        v16h& kf = c ? kf1 : kf0;
#pragma unroll
        for (int e = 0; e < 8; e++) { kf[e] = lo[e]; kf[8 + e] = hh[e]; }
      }
      v8f z = wmma_f16(qf[0], kf0, vzero);
      s[jt] = wmma_f16(qf[1], kf1, z);
    }
    if (mask) {
#pragma unroll
      for (int jt = 0; jt < 2; jt++) {
        const int kcol = kb0 + jt * 16 + ln;
#pragma unroll
        for (int r = 0; r < 8; r++) {
          const int qr = q0 + r + 8 * hi;
          s[jt][r] += mask[(size_t)qr * kS + kcol];
        }
      }
    }

    // ---- online softmax update (row = r + 8*hi across lanes ln=0..15) ----
    float p0[8], p1[8], alpha[8];
#pragma unroll
    for (int r = 0; r < 8; r++) {
      float bm = fmaxf(s[0][r], s[1][r]);
#pragma unroll
      for (int m = 1; m < 16; m <<= 1) bm = fmaxf(bm, __shfl_xor(bm, m, 32));
      const float nm = fmaxf(mrow[r], bm);
      const float al = __expf(mrow[r] - nm);
      mrow[r] = nm;
      const float e0 = __expf(s[0][r] - nm);
      const float e1 = __expf(s[1][r] - nm);
      float rs = e0 + e1;
#pragma unroll
      for (int m = 1; m < 16; m <<= 1) rs += __shfl_xor(rs, m, 32);
      lrow[r] = lrow[r] * al + rs;
      p0[r] = e0; p1[r] = e1; alpha[r] = al;
    }
#pragma unroll
    for (int nt = 0; nt < 4; nt++)
#pragma unroll
      for (int r = 0; r < 8; r++) ctx[nt][r] *= alpha[r];

    // ---- transpose P (C-layout -> A-layout) via per-wave LDS ----
#pragma unroll
    for (int r = 0; r < 8; r++) {
      const int rr = r + 8 * hi;
      Pl[wv][rr * 40 + ln]      = (_Float16)p0[r];
      Pl[wv][rr * 40 + 16 + ln] = (_Float16)p1[r];
    }
    asm volatile("s_wait_dscnt 0x0" ::: "memory");
    v16h pf;
    {
      const int c0 = hi ? 8 : 0;
      v8h lo = *reinterpret_cast<const v8h*>(&Pl[wv][ln * 40 + c0]);
      v8h hh = *reinterpret_cast<const v8h*>(&Pl[wv][ln * 40 + c0 + 16]);
#pragma unroll
      for (int e = 0; e < 8; e++) { pf[e] = lo[e]; pf[8 + e] = hh[e]; }
    }

    // ---- ctx += P @ V (V fragments from transposed LDS tile) ----
#pragma unroll
    for (int nt = 0; nt < 4; nt++) {
      v16h vf;
      const int d  = nt * 16 + ln;
      const int j0 = hi ? 8 : 0;
      v8h lo = *reinterpret_cast<const v8h*>(&Vt[d * 40 + j0]);
      v8h hh = *reinterpret_cast<const v8h*>(&Vt[d * 40 + j0 + 16]);
#pragma unroll
      for (int e = 0; e < 8; e++) { vf[e] = lo[e]; vf[8 + e] = hh[e]; }
      ctx[nt] = wmma_f16(pf, vf, ctx[nt]);
    }
    __syncthreads();   // protect Ks/Vt before next block-step restage
  }

  // ---- normalize and scatter into ctx buffer [B*S, D] at head slice ----
#pragma unroll
  for (int nt = 0; nt < 4; nt++) {
#pragma unroll
    for (int r = 0; r < 8; r++) {
      const int qr = q0 + r + 8 * hi;
      const float inv = 1.0f / lrow[r];
      O[((size_t)b * kS + qr) * kD + (size_t)h * kDH + nt * 16 + ln] =
          ctx[nt][r] * inv;
    }
  }
}

// ---------------------------------------------------------------------------
// LayerNorm over D=1024, one block per row, 4 elements/thread.
// ---------------------------------------------------------------------------
__global__ void __launch_bounds__(256)
ln_kernel(const float* __restrict__ x, const float* __restrict__ g,
          const float* __restrict__ bta, float* __restrict__ out) {
  const int row = blockIdx.x;
  const int t   = threadIdx.x;
  const float4 v = reinterpret_cast<const float4*>(x + (size_t)row * kD)[t];
  __shared__ float red[256];
  red[t] = v.x + v.y + v.z + v.w;
  __syncthreads();
  for (int o = 128; o > 0; o >>= 1) {
    if (t < o) red[t] += red[t + o];
    __syncthreads();
  }
  const float mu = red[0] * (1.0f / kD);
  __syncthreads();
  const float d0 = v.x - mu, d1 = v.y - mu, d2 = v.z - mu, d3 = v.w - mu;
  red[t] = d0 * d0 + d1 * d1 + d2 * d2 + d3 * d3;
  __syncthreads();
  for (int o = 128; o > 0; o >>= 1) {
    if (t < o) red[t] += red[t + o];
    __syncthreads();
  }
  const float rstd = rsqrtf(red[0] * (1.0f / kD) + 1e-5f);
  const float4 gg = reinterpret_cast<const float4*>(g)[t];
  const float4 bb = reinterpret_cast<const float4*>(bta)[t];
  float4 o4 = {d0 * rstd * gg.x + bb.x, d1 * rstd * gg.y + bb.y,
               d2 * rstd * gg.z + bb.z, d3 * rstd * gg.w + bb.w};
  reinterpret_cast<float4*>(out + (size_t)row * kD)[t] = o4;
}

// ---------------------------------------------------------------------------
// Host orchestration
// ---------------------------------------------------------------------------
enum {
  IN_X = 0, IN_MEM, IN_MASK,
  P_SA_WQ, P_SA_BQ, P_SA_WK, P_SA_BK, P_SA_WV, P_SA_BV, P_SA_WO, P_SA_BO,
  P_MA_WQ, P_MA_BQ, P_MA_WK, P_MA_BK, P_MA_WV, P_MA_BV, P_MA_WO, P_MA_BO,
  P_LN1_G, P_LN1_B, P_LN2_G, P_LN2_B, P_LN3_G, P_LN3_B,
  P_FF_W1, P_FF_B1, P_FF_W2, P_FF_B2, P_LNF_G, P_LNF_B
};

extern "C" void kernel_launch(void* const* d_in, const int* in_sizes, int n_in,
                              void* d_out, int out_size, void* d_ws,
                              size_t ws_size, hipStream_t stream) {
  (void)in_sizes; (void)n_in; (void)out_size; (void)ws_size;

  const size_t nAct = (size_t)kRows * kD;      // 4M floats
  float* xb   = (float*)d_ws;                  // running activations
  float* hb   = xb + nAct;                     // LN output
  float* qb   = hb + nAct;
  float* kbuf = qb + nAct;
  float* vbuf = kbuf + nAct;
  float* cbuf = vbuf + nAct;                   // attention context
  float* fbuf = cbuf + nAct;                   // FFN hidden [4096 x 4096]
  _Float16* wt16 = (_Float16*)(fbuf + (size_t)kRows * kF);  // f16 W^T scratch

  const float* x_in  = (const float*)d_in[IN_X];
  const float* mem   = (const float*)d_in[IN_MEM];
  const float* amask = (const float*)d_in[IN_MASK];

  hipMemcpyAsync(xb, x_in, nAct * sizeof(float), hipMemcpyDeviceToDevice,
                 stream);

  // convert W[K,N] fp32 -> wt16[N,K] f16, then WMMA GEMM against it
  auto gemm = [&](const float* Ap, const float* Wp, const float* bp,
                  const float* rp, float* Cp, int Mr, int K, int N, int relu) {
    dim3 gcv(N / 32, K / 32);
    wcvt_t<<<gcv, 256, 0, stream>>>(Wp, wt16, K, N);
    dim3 grid(N / 128, Mr / 128);
    if (rp)
      gemm_f16<true, false><<<grid, 256, 0, stream>>>(Ap, wt16, bp, rp, Cp, Mr, K, N);
    else if (relu)
      gemm_f16<false, true><<<grid, 256, 0, stream>>>(Ap, wt16, bp, nullptr, Cp, Mr, K, N);
    else
      gemm_f16<false, false><<<grid, 256, 0, stream>>>(Ap, wt16, bp, nullptr, Cp, Mr, K, N);
  };
  auto ln = [&](const float* xp, const float* gp, const float* bp, float* op) {
    ln_kernel<<<kRows, 256, 0, stream>>>(xp, gp, bp, op);
  };
  auto attn = [&](const float* Qp, const float* Kp, const float* Vp,
                  const float* mp, float* Op, int Skv) {
    dim3 grid(kS / 128, kB * kH);
    attn_flash<<<grid, 256, 0, stream>>>(Qp, Kp, Vp, mp, Op, Skv);
  };
  auto P = [&](int idx, size_t per, int layer) {
    return (const float*)d_in[idx] + (size_t)layer * per;
  };

  const size_t WDD = (size_t)kD * kD, BD = kD, WDF = (size_t)kD * kF, BF = kF;

  for (int i = 0; i < 6; i++) {
    // ---- self-attention sublayer ----
    ln(xb, P(P_LN1_G, BD, i), P(P_LN1_B, BD, i), hb);
    gemm(hb, P(P_SA_WQ, WDD, i), P(P_SA_BQ, BD, i), nullptr, qb,   kRows, kD, kD, 0);
    gemm(hb, P(P_SA_WK, WDD, i), P(P_SA_BK, BD, i), nullptr, kbuf, kRows, kD, kD, 0);
    gemm(hb, P(P_SA_WV, WDD, i), P(P_SA_BV, BD, i), nullptr, vbuf, kRows, kD, kD, 0);
    attn(qb, kbuf, vbuf, amask, cbuf, kS);
    gemm(cbuf, P(P_SA_WO, WDD, i), P(P_SA_BO, BD, i), xb, xb, kRows, kD, kD, 0);

    // ---- cross-attention sublayer ----
    ln(xb, P(P_LN2_G, BD, i), P(P_LN2_B, BD, i), hb);
    gemm(hb,  P(P_MA_WQ, WDD, i), P(P_MA_BQ, BD, i), nullptr, qb,   kRows, kD, kD, 0);
    gemm(mem, P(P_MA_WK, WDD, i), P(P_MA_BK, BD, i), nullptr, kbuf, kRows, kD, kD, 0);
    gemm(mem, P(P_MA_WV, WDD, i), P(P_MA_BV, BD, i), nullptr, vbuf, kRows, kD, kD, 0);
    attn(qb, kbuf, vbuf, nullptr, cbuf, kS);
    gemm(cbuf, P(P_MA_WO, WDD, i), P(P_MA_BO, BD, i), xb, xb, kRows, kD, kD, 0);

    // ---- feed-forward sublayer ----
    ln(xb, P(P_LN3_G, BD, i), P(P_LN3_B, BD, i), hb);
    gemm(hb,   P(P_FF_W1, WDF, i), P(P_FF_B1, BF, i), nullptr, fbuf, kRows, kD, kF, 1);
    gemm(fbuf, P(P_FF_W2, WDF, i), P(P_FF_B2, BD, i), xb, xb,        kRows, kF, kD, 0);
  }

  // final layernorm -> output
  ln(xb, (const float*)d_in[P_LNF_G], (const float*)d_in[P_LNF_B],
     (float*)d_out);
}
